// TransformerEncoderLayer_14087492731499
// MI455X (gfx1250) — matrix operand order repaired and checked by reading, not compiled, not run on hardware
//
#include <hip/hip_runtime.h>
#include <hip/hip_bf16.h>

typedef unsigned short u16;
typedef unsigned int   u32;
typedef unsigned char  u8;
typedef __attribute__((ext_vector_type(16))) __bf16 v16bf;
typedef __attribute__((ext_vector_type(8)))  float  v8f;

#define ROWS   12608          // B*N = 64*197
#define NSEQ   197
#define EMB    640
#define NHEAD  10
#define HD     64
#define DFF2   2560
#define NPM    208            // 13*16  padded n rows
#define NPK    224            // 14*16  padded m cols (7 k-chunks of 32)

// ---------------- helpers ----------------
__device__ __forceinline__ u16 f32_to_bf16(float f) {
  union { float f; u32 u; } c; c.f = f;
  return (u16)((c.u + 0x7FFFu + ((c.u >> 16) & 1u)) >> 16);
}

// A-fragment (16x32 bf16): lane = row M (lane&15), half = lane>>4.
// elements j: k = half*16 + j  -> two contiguous 16B runs.
__device__ __forceinline__ v16bf load_a_frag_bf16(const u16* rowbase, int kc, int lhalf) {
  union { v16bf v; uint4 q[2]; } cv;
  cv.q[0] = *(const uint4*)(rowbase + kc * 32 + lhalf * 16);
  cv.q[1] = *(const uint4*)(rowbase + kc * 32 + lhalf * 16 + 8);
  return cv.v;
}

// Same A layout, but source is fp32; convert to bf16 (RNE) on the fly.
__device__ __forceinline__ v16bf load_a_frag_f32(const float* rowbase, int kc, int lhalf) {
  const float* p0 = rowbase + kc * 32 + lhalf * 16;
  const float* p1 = p0 + 8;
  union { v16bf v; u32 u[8]; } cv;
#pragma unroll
  for (int i = 0; i < 4; i++)
    cv.u[i] = (u32)f32_to_bf16(p0[2 * i]) | ((u32)f32_to_bf16(p0[2 * i + 1]) << 16);
#pragma unroll
  for (int i = 0; i < 4; i++)
    cv.u[4 + i] = (u32)f32_to_bf16(p1[2 * i]) | ((u32)f32_to_bf16(p1[2 * i + 1]) << 16);
  return cv.v;
}

// B-fragment (32x16 bf16): lane = col N (lane&15); k = kc*32 + half*16 + j
// -> one contiguous 32B run per lane from row-major [N][K] storage of B^T.
__device__ __forceinline__ v16bf load_b_frag_bf16(const u16* rowbase, int koff) {
  union { v16bf v; uint4 q[2]; } cv;
  cv.q[0] = *(const uint4*)(rowbase + koff);
  cv.q[1] = *(const uint4*)(rowbase + koff + 8);
  return cv.v;
}

#define WMMA_BF16(A, B, C) \
  __builtin_amdgcn_wmma_f32_16x16x32_bf16(false, (A), false, (B), (short)0, (C), false, false)

// ---------------- small prep kernels ----------------
__global__ void cvt_bf16_kernel(const float* __restrict__ in, u16* __restrict__ out, int n) {
  for (int i = blockIdx.x * blockDim.x + threadIdx.x; i < n; i += gridDim.x * blockDim.x)
    out[i] = f32_to_bf16(in[i]);
}

// RPE tables: pad 30 rows -> 32 (rows 30,31 = 0). K-tables row-major [32][64];
// V-tables transposed [64][32] so B-fragments are contiguous.
__global__ void prep_tabs_kernel(const float* kv, const float* kh, const float* vv, const float* vh,
                                 u16* tKv, u16* tKh, u16* tVvT, u16* tVhT) {
  int t = blockIdx.x * blockDim.x + threadIdx.x;
  if (t >= 32 * 64) return;
  int r = t >> 6, d = t & 63;
  tKv[r * 64 + d]  = (r < 30) ? f32_to_bf16(kv[r * 64 + d]) : (u16)0;
  tKh[r * 64 + d]  = (r < 30) ? f32_to_bf16(kh[r * 64 + d]) : (u16)0;
  tVvT[d * 32 + r] = (r < 30) ? f32_to_bf16(vv[r * 64 + d]) : (u16)0;
  tVhT[d * 32 + r] = (r < 30) ? f32_to_bf16(vh[r * 64 + d]) : (u16)0;
}

__global__ void prep_idx_kernel(u8* __restrict__ iv, u8* __restrict__ ih) {
  for (int t = blockIdx.x * blockDim.x + threadIdx.x; t < NSEQ * NSEQ;
       t += gridDim.x * blockDim.x) {
    int n = t / NSEQ, m = t % NSEQ;
    u8 a = 0, b = 0;
    if (n > 0 && m > 0) {
      int i = n - 1, j = m - 1;
      int dv = j / 14 - i / 14;
      int dh = j % 14 - i % 14;
      dv = dv < -14 ? -14 : (dv > 14 ? 14 : dv);
      dh = dh < -14 ? -14 : (dh > 14 ? 14 : dh);
      a = (u8)(dv + 15); b = (u8)(dh + 15);
    }
    iv[t] = a; ih[t] = b;
  }
}

// ---------------- LayerNorm -> bf16 (one wave per row, E=640) ----------------
__global__ __launch_bounds__(256) void layernorm_bf16_kernel(
    const float* __restrict__ x, const float* __restrict__ g, const float* __restrict__ bta,
    u16* __restrict__ out, int rows) {
  int wid = threadIdx.x >> 5, lane = threadIdx.x & 31;
  int row = blockIdx.x * 8 + wid;
  if (row >= rows) return;
  const float* xr = x + (size_t)row * EMB;
  float vals[20];
  float s = 0.f;
#pragma unroll
  for (int i = 0; i < 20; i++) { vals[i] = xr[lane + 32 * i]; s += vals[i]; }
#pragma unroll
  for (int o = 16; o >= 1; o >>= 1) s += __shfl_xor(s, o, 32);
  float mu = s * (1.0f / EMB);
  float vs = 0.f;
#pragma unroll
  for (int i = 0; i < 20; i++) { float d = vals[i] - mu; vs += d * d; }
#pragma unroll
  for (int o = 16; o >= 1; o >>= 1) vs += __shfl_xor(vs, o, 32);
  float rinv = rsqrtf(vs * (1.0f / EMB) + 1e-5f);
  u16* orow = out + (size_t)row * EMB;
#pragma unroll
  for (int i = 0; i < 20; i++) {
    int c = lane + 32 * i;
    orow[c] = f32_to_bf16((vals[i] - mu) * rinv * g[c] + bta[c]);
  }
}

// ---------------- generic WMMA GEMM: C = A[M,K](bf16) @ W[N,K]^T(bf16) ----------------
// block = 256 threads (8 waves); block tile 256(M) x 64(N);
// each wave: 32 rows x 64 cols = 2 A-frags, 4 B-frags, 8 WMMAs per K-step.
__global__ __launch_bounds__(256) void gemm_bf16_kernel(
    const u16* __restrict__ A, const u16* __restrict__ W,
    const float* __restrict__ bias, const float* __restrict__ residual,
    float* __restrict__ outF, u16* __restrict__ outB,
    int M, int N, int K, int act) {
  int lane = threadIdx.x & 31, wid = threadIdx.x >> 5;
  int lmod = lane & 15, lhalf = lane >> 4;
  int m0 = blockIdx.x * 256 + wid * 32;
  int n0 = blockIdx.y * 64;
  int a_r0 = m0 + lmod;      if (a_r0 >= M) a_r0 = M - 1;
  int a_r1 = m0 + 16 + lmod; if (a_r1 >= M) a_r1 = M - 1;
  const u16* arow0 = A + (size_t)a_r0 * K;
  const u16* arow1 = A + (size_t)a_r1 * K;
  const u16* wr0 = W + (size_t)(n0 +  0 + lmod) * K;
  const u16* wr1 = W + (size_t)(n0 + 16 + lmod) * K;
  const u16* wr2 = W + (size_t)(n0 + 32 + lmod) * K;
  const u16* wr3 = W + (size_t)(n0 + 48 + lmod) * K;
  v8f acc[2][4]; v8f z = {};
#pragma unroll
  for (int i = 0; i < 2; i++)
#pragma unroll
    for (int t = 0; t < 4; t++) acc[i][t] = z;
  int ksteps = K >> 5;
  for (int kc = 0; kc < ksteps; kc++) {
    v16bf a0 = load_a_frag_bf16(arow0, kc, lhalf);
    v16bf a1 = load_a_frag_bf16(arow1, kc, lhalf);
    int koff = kc * 32 + lhalf * 16;
    v16bf b0 = load_b_frag_bf16(wr0, koff);
    v16bf b1 = load_b_frag_bf16(wr1, koff);
    v16bf b2 = load_b_frag_bf16(wr2, koff);
    v16bf b3 = load_b_frag_bf16(wr3, koff);
    acc[0][0] = WMMA_BF16(a0, b0, acc[0][0]);
    acc[0][1] = WMMA_BF16(a0, b1, acc[0][1]);
    acc[0][2] = WMMA_BF16(a0, b2, acc[0][2]);
    acc[0][3] = WMMA_BF16(a0, b3, acc[0][3]);
    acc[1][0] = WMMA_BF16(a1, b0, acc[1][0]);
    acc[1][1] = WMMA_BF16(a1, b1, acc[1][1]);
    acc[1][2] = WMMA_BF16(a1, b2, acc[1][2]);
    acc[1][3] = WMMA_BF16(a1, b3, acc[1][3]);
  }
#pragma unroll
  for (int i = 0; i < 2; i++) {
#pragma unroll
    for (int t = 0; t < 4; t++) {
      int col = n0 + t * 16 + lmod;
      float bv = bias ? bias[col] : 0.f;
#pragma unroll
      for (int r = 0; r < 8; r++) {
        int row = m0 + i * 16 + r + 8 * lhalf;
        if (row < M) {
          float f = acc[i][t][r] + bv;
          if (act) f = 0.5f * f * (1.0f + erff(f * 0.70710678118f));  // exact GELU
          if (residual) f += residual[(size_t)row * N + col];
          if (outF) outF[(size_t)row * N + col] = f;
          if (outB) outB[(size_t)row * N + col] = f32_to_bf16(f);
        }
      }
    }
  }
}

// ---------------- fused attention: one workgroup per (b,h) ----------------
// LDS layout (316 KB, fits CDNA5's 320 KB/WG):
//   sS   fp32 [208][224]   scores -> attn probs
//   sQ   bf16 [208][64]
//   sK   bf16 [224][64]
//   sVt  bf16 [64][224]    V transposed (B-matrix friendly)
//   sPA  fp32 [2][208][32] P_v/P_h (q . rpe table), reused as A_v/A_h bucket sums
#define S_OFF   0
#define Q_OFF   186368
#define K_OFF   212992
#define V_OFF   241664
#define PA_OFF  270336
#define ATT_SMEM 323584

__global__ __launch_bounds__(256, 1) void attention_kernel(
    const u16* __restrict__ q, const u16* __restrict__ k, const u16* __restrict__ v,
    const u16* __restrict__ tKv, const u16* __restrict__ tKh,
    const u16* __restrict__ tVvT, const u16* __restrict__ tVhT,
    const u8* __restrict__ idxv, const u8* __restrict__ idxh,
    u16* __restrict__ o) {
  extern __shared__ char smem[];
  float* sS  = (float*)(smem + S_OFF);
  u16*   sQ  = (u16*)(smem + Q_OFF);
  u16*   sK  = (u16*)(smem + K_OFF);
  u16*   sVt = (u16*)(smem + V_OFF);
  float* sPA = (float*)(smem + PA_OFF);   // [2][208][32]

  int tid = threadIdx.x;
  int lane = tid & 31, wid = tid >> 5;
  int lmod = lane & 15, lhalf = lane >> 4;
  int bh = blockIdx.x;
  int b = bh / NHEAD, h = bh % NHEAD;
  const u16* qg = q + (size_t)b * NSEQ * EMB + h * HD;
  const u16* kg = k + (size_t)b * NSEQ * EMB + h * HD;
  const u16* vg = v + (size_t)b * NSEQ * EMB + h * HD;

  // ---- load Q,K (row-major), V (transposed) into LDS; zero-pad ----
  for (int i = tid; i < NPM * 8; i += 256) {      // 8 bf16 (16B) per chunk
    int r = i >> 3, c = (i & 7) * 8;
    uint4 val = {0, 0, 0, 0};
    if (r < NSEQ) val = *(const uint4*)(qg + (size_t)r * EMB + c);
    *(uint4*)(sQ + r * 64 + c) = val;
  }
  for (int i = tid; i < NPK * 8; i += 256) {
    int r = i >> 3, c = (i & 7) * 8;
    uint4 val = {0, 0, 0, 0};
    if (r < NSEQ) val = *(const uint4*)(kg + (size_t)r * EMB + c);
    *(uint4*)(sK + r * 64 + c) = val;
  }
  for (int i = tid; i < NPK * 64; i += 256) {
    int m = i >> 6, d = i & 63;
    sVt[d * NPK + m] = (m < NSEQ) ? vg[(size_t)m * EMB + d] : (u16)0;
  }
  __syncthreads();

  // ---- S = Q @ K^T : 13x14 tiles of 16x16, K-dim 64 (2 wmma) ----
  for (int t = wid; t < 13 * 14; t += 8) {
    int mt = t / 14, nt = t % 14;
    v8f acc = {};
    const u16* arow = sQ + (mt * 16 + lmod) * 64;
    const u16* brow = sK + (nt * 16 + lmod) * 64;
#pragma unroll
    for (int kc = 0; kc < 2; kc++) {
      v16bf a = load_a_frag_bf16(arow, kc, lhalf);
      v16bf bb = load_b_frag_bf16(brow, kc * 32 + lhalf * 16);
      acc = WMMA_BF16(a, bb, acc);
    }
#pragma unroll
    for (int r = 0; r < 8; r++)
      sS[(mt * 16 + r + 8 * lhalf) * NPK + nt * 16 + lmod] = acc[r];
  }

  // ---- P_v/P_h = Q @ tab^T : 13 mt x 2 t-tiles x 2 tables ----
  for (int t = wid; t < 52; t += 8) {
    int which = t & 1, tt = (t >> 1) & 1, mt = t >> 2;
    const u16* tab = which ? tKh : tKv;
    v8f acc = {};
    const u16* arow = sQ + (mt * 16 + lmod) * 64;
    const u16* brow = tab + (tt * 16 + lmod) * 64;
#pragma unroll
    for (int kc = 0; kc < 2; kc++) {
      v16bf a = load_a_frag_bf16(arow, kc, lhalf);
      v16bf bb = load_b_frag_bf16(brow, kc * 32 + lhalf * 16);
      acc = WMMA_BF16(a, bb, acc);
    }
    float* dst = sPA + which * (NPM * 32);
#pragma unroll
    for (int r = 0; r < 8; r++)
      dst[(mt * 16 + r + 8 * lhalf) * 32 + tt * 16 + lmod] = acc[r];
  }
  __syncthreads();

  // ---- per-row: add RPE via P gather, scale, softmax, bucket-sum A_v/A_h ----
  const float scale = 0.125f;  // 64^-0.5
  for (int n = wid; n < NPM; n += 8) {
    if (n >= NSEQ) {
      for (int m = lane; m < NPK; m += 32) sS[n * NPK + m] = 0.f;
      sPA[n * 32 + lane] = 0.f;
      sPA[NPM * 32 + n * 32 + lane] = 0.f;
      continue;
    }
    const float* pv = sPA + n * 32;
    const float* ph = sPA + NPM * 32 + n * 32;
    const u8* ivr = idxv + n * NSEQ;
    const u8* ihr = idxh + n * NSEQ;
    float sv[7];
    float mx = -3.4e38f;
#pragma unroll
    for (int j = 0; j < 7; j++) {
      int m = lane + 32 * j;
      float s = -3.4e38f;
      if (m < NSEQ)
        s = (sS[n * NPK + m] + pv[ivr[m]] + ph[ihr[m]]) * scale;
      sv[j] = s;
      mx = fmaxf(mx, s);
    }
#pragma unroll
    for (int off = 16; off >= 1; off >>= 1) mx = fmaxf(mx, __shfl_xor(mx, off, 32));
    float sum = 0.f;
#pragma unroll
    for (int j = 0; j < 7; j++) {
      int m = lane + 32 * j;
      float e = (m < NSEQ) ? __expf(sv[j] - mx) : 0.f;
      sv[j] = e; sum += e;
    }
#pragma unroll
    for (int off = 16; off >= 1; off >>= 1) sum += __shfl_xor(sum, off, 32);
    float inv = 1.0f / sum;
    // P row no longer needed -> reuse as bucket accumulators (same wave, in-order LDS)
    sPA[n * 32 + lane] = 0.f;
    sPA[NPM * 32 + n * 32 + lane] = 0.f;
    float* av = sPA + n * 32;
    float* ah = sPA + NPM * 32 + n * 32;
#pragma unroll
    for (int j = 0; j < 7; j++) {
      int m = lane + 32 * j;
      float a = sv[j] * inv;
      if (m < NSEQ) {
        sS[n * NPK + m] = a;
        atomicAdd(&av[ivr[m]], a);
        atomicAdd(&ah[ihr[m]], a);
      } else {
        sS[n * NPK + m] = 0.f;
      }
    }
  }
  __syncthreads();

  // ---- O = attn @ V + A_v @ tabV_v + A_h @ tabV_h : 13x4 tiles ----
  for (int t = wid; t < 13 * 4; t += 8) {
    int mt = t >> 2, nt = t & 3;
    v8f acc = {};
    const float* arow = sS + (mt * 16 + lmod) * NPK;
    const u16* brow = sVt + (nt * 16 + lmod) * NPK;
#pragma unroll
    for (int kc = 0; kc < 7; kc++) {
      v16bf a = load_a_frag_f32(arow, kc, lhalf);
      v16bf bb = load_b_frag_bf16(brow, kc * 32 + lhalf * 16);
      acc = WMMA_BF16(a, bb, acc);
    }
    {
      v16bf a = load_a_frag_f32(sPA + (mt * 16 + lmod) * 32, 0, lhalf);
      v16bf bb = load_b_frag_bf16(tVvT + (nt * 16 + lmod) * 32, lhalf * 16);
      acc = WMMA_BF16(a, bb, acc);
      a = load_a_frag_f32(sPA + NPM * 32 + (mt * 16 + lmod) * 32, 0, lhalf);
      bb = load_b_frag_bf16(tVhT + (nt * 16 + lmod) * 32, lhalf * 16);
      acc = WMMA_BF16(a, bb, acc);
    }
#pragma unroll
    for (int r = 0; r < 8; r++) {
      int row = mt * 16 + r + 8 * lhalf;
      if (row < NSEQ) {
        int d = nt * 16 + lmod;
        o[((size_t)b * NSEQ + row) * EMB + h * HD + d] = f32_to_bf16(acc[r]);
      }
    }
  }
}

// ---------------- host orchestration ----------------
extern "C" void kernel_launch(void* const* d_in, const int* in_sizes, int n_in,
                              void* d_out, int out_size, void* d_ws, size_t ws_size,
                              hipStream_t stream) {
  const float* x    = (const float*)d_in[0];
  const float* ln1g = (const float*)d_in[1];
  const float* ln1b = (const float*)d_in[2];
  const float* Wq   = (const float*)d_in[3];
  const float* Wk   = (const float*)d_in[4];
  const float* Wv   = (const float*)d_in[5];
  const float* Wp   = (const float*)d_in[6];
  const float* bp   = (const float*)d_in[7];
  const float* rpkv = (const float*)d_in[8];
  const float* rpkh = (const float*)d_in[9];
  const float* rpvv = (const float*)d_in[10];
  const float* rpvh = (const float*)d_in[11];
  const float* ln2g = (const float*)d_in[12];
  const float* ln2b = (const float*)d_in[13];
  const float* W1   = (const float*)d_in[14];
  const float* b1   = (const float*)d_in[15];
  const float* W2   = (const float*)d_in[16];
  const float* b2   = (const float*)d_in[17];
  float* out = (float*)d_out;

  char* ws = (char*)d_ws;
  size_t off = 0;
  auto alloc = [&](size_t bytes) -> char* {
    char* p = ws + off;
    off += (bytes + 255) & ~(size_t)255;
    return p;
  };

  u16* h1  = (u16*)alloc((size_t)ROWS * EMB * 2);
  u16* qb  = (u16*)alloc((size_t)ROWS * EMB * 2);
  u16* kb  = (u16*)alloc((size_t)ROWS * EMB * 2);
  u16* vb  = (u16*)alloc((size_t)ROWS * EMB * 2);
  u16* ob  = (u16*)alloc((size_t)ROWS * EMB * 2);
  float* x1 = (float*)alloc((size_t)ROWS * EMB * 4);
  u16* h2  = (u16*)alloc((size_t)ROWS * EMB * 2);
  u16* gb  = (u16*)alloc((size_t)ROWS * DFF2 * 2);
  u16* wqb = (u16*)alloc((size_t)EMB * EMB * 2);
  u16* wkb = (u16*)alloc((size_t)EMB * EMB * 2);
  u16* wvb = (u16*)alloc((size_t)EMB * EMB * 2);
  u16* wpb = (u16*)alloc((size_t)EMB * EMB * 2);
  u16* w1b = (u16*)alloc((size_t)DFF2 * EMB * 2);
  u16* w2b = (u16*)alloc((size_t)EMB * DFF2 * 2);
  u16* tKv  = (u16*)alloc(32 * 64 * 2);
  u16* tKh  = (u16*)alloc(32 * 64 * 2);
  u16* tVvT = (u16*)alloc(64 * 32 * 2);
  u16* tVhT = (u16*)alloc(64 * 32 * 2);
  u8* idxv = (u8*)alloc(NSEQ * NSEQ);
  u8* idxh = (u8*)alloc(NSEQ * NSEQ);

  // weight / table / index prep
  cvt_bf16_kernel<<<1024, 256, 0, stream>>>(Wq, wqb, EMB * EMB);
  cvt_bf16_kernel<<<1024, 256, 0, stream>>>(Wk, wkb, EMB * EMB);
  cvt_bf16_kernel<<<1024, 256, 0, stream>>>(Wv, wvb, EMB * EMB);
  cvt_bf16_kernel<<<1024, 256, 0, stream>>>(Wp, wpb, EMB * EMB);
  cvt_bf16_kernel<<<1024, 256, 0, stream>>>(W1, w1b, DFF2 * EMB);
  cvt_bf16_kernel<<<1024, 256, 0, stream>>>(W2, w2b, EMB * DFF2);
  prep_tabs_kernel<<<8, 256, 0, stream>>>(rpkv, rpkh, rpvv, rpvh, tKv, tKh, tVvT, tVhT);
  prep_idx_kernel<<<152, 256, 0, stream>>>(idxv, idxh);

  dim3 blk(256);
  dim3 g640((ROWS + 255) / 256, EMB / 64);
  dim3 g2560((ROWS + 255) / 256, DFF2 / 64);

  // LN1 -> h1 (bf16)
  layernorm_bf16_kernel<<<(ROWS + 7) / 8, blk, 0, stream>>>(x, ln1g, ln1b, h1, ROWS);

  // Q, K, V projections
  gemm_bf16_kernel<<<g640, blk, 0, stream>>>(h1, wqb, nullptr, nullptr, nullptr, qb,
                                             ROWS, EMB, EMB, 0);
  gemm_bf16_kernel<<<g640, blk, 0, stream>>>(h1, wkb, nullptr, nullptr, nullptr, kb,
                                             ROWS, EMB, EMB, 0);
  gemm_bf16_kernel<<<g640, blk, 0, stream>>>(h1, wvb, nullptr, nullptr, nullptr, vb,
                                             ROWS, EMB, EMB, 0);

  // fused attention with 2D RPE, one WG per (b,h)
  attention_kernel<<<64 * NHEAD, blk, ATT_SMEM, stream>>>(qb, kb, vb, tKv, tKh, tVvT, tVhT,
                                                          idxv, idxh, ob);

  // x1 = x + o @ Wp^T + bp
  gemm_bf16_kernel<<<g640, blk, 0, stream>>>(ob, wpb, bp, x, x1, nullptr,
                                             ROWS, EMB, EMB, 0);

  // LN2 -> h2 (bf16)
  layernorm_bf16_kernel<<<(ROWS + 7) / 8, blk, 0, stream>>>(x1, ln2g, ln2b, h2, ROWS);

  // g = gelu(h2 @ W1^T + b1)
  gemm_bf16_kernel<<<g2560, blk, 0, stream>>>(h2, w1b, b1, nullptr, nullptr, gb,
                                              ROWS, DFF2, EMB, 1);

  // out = x1 + g @ W2^T + b2
  gemm_bf16_kernel<<<g640, blk, 0, stream>>>(gb, w2b, b2, x1, out, nullptr,
                                             ROWS, EMB, DFF2, 0);
}
